// _Emotion2vecModel_28707561406479
// MI455X (gfx1250) — matrix operand order repaired and checked by reading, not compiled, not run on hardware
//
#include <hip/hip_runtime.h>
#include <cstdint>
#include <cstddef>

// ---------------------------------------------------------------------------
// CDNA5 (gfx1250) wave32 WMMA implementation of the emotion2vec forward pass.
// All GEMM-shaped work (conv im2col GEMMs, qkv, attention scores, att@v,
// out-proj, MLP) runs on v_wmma_f32_16x16x32_bf16. f32 accumulate, bf16 data.
// Each wave computes a 32(M)x64(N) tile: 8 WMMAs per 32-wide K step, which
// doubles operand reuse versus a 16x64 tile and halves vmem/wmma ratio.
// ---------------------------------------------------------------------------

typedef __attribute__((ext_vector_type(16))) __bf16 bf16x16;
typedef __attribute__((ext_vector_type(8)))  __bf16 bf16x8;
typedef __attribute__((ext_vector_type(8)))  float  f32x8;

#define DEV __device__ __forceinline__

DEV float gelu_f(float x) { return 0.5f * x * (1.0f + erff(x * 0.7071067811865476f)); }

// A-fragment (16x32 bf16): lane L -> row = L&15, kb = (L>>4)*8,
// elements 0..7 = K[kb..kb+7], elements 8..15 = K[kb+16..kb+23]
DEV bf16x16 load_afrag(const __bf16* p) {
  bf16x8 lo = *(const bf16x8*)p;
  bf16x8 hi = *(const bf16x8*)(p + 16);
  bf16x16 r;
#pragma unroll
  for (int i = 0; i < 8; ++i) { r[i] = lo[i]; r[i + 8] = hi[i]; }
  return r;
}
// B-fragment (32x16 bf16): lane L -> col = L&15, kb = (L>>4)*8,
// elements 0..7 = K[kb..kb+7], elements 8..15 = K[kb+16..kb+23]  (Wt stored [N][K])
DEV bf16x16 load_bfrag(const __bf16* p) {
  bf16x8 lo = *(const bf16x8*)p;
  bf16x8 hi = *(const bf16x8*)(p + 16);
  bf16x16 r;
#pragma unroll
  for (int i = 0; i < 8; ++i) { r[i] = lo[i]; r[i + 8] = hi[i]; }
  return r;
}

DEV f32x8 wmma_bf16(bf16x16 a, bf16x16 b, f32x8 c) {
  return __builtin_amdgcn_wmma_f32_16x16x32_bf16(false, a, false, b, (short)0, c,
                                                 false, false);
}

DEV float wave_sum(float v) {
#pragma unroll
  for (int off = 16; off > 0; off >>= 1) v += __shfl_xor(v, off, 32);
  return v;
}
DEV float wave_max(float v) {
#pragma unroll
  for (int off = 16; off > 0; off >>= 1) v = fmaxf(v, __shfl_xor(v, off, 32));
  return v;
}

// ---------------------------------------------------------------------------
// Generic WMMA GEMM: C[M][N] = A[M][K](bf16) @ Wt[N][K](bf16, pre-transposed)
// One wave per block computing a 32(M) x 64(N) tile (2x4 accumulators).
// Requires M % 32 == 0, N % 64 == 0, K % 32 == 0 (all call sites satisfy).
// Epilogues: 0=store f32, 1=+bias, 2=+bias residual-add into Cf,
//            3=+bias gelu -> bf16, 4=store 2*(acc+bias) (the "m+m" quirk).
// ---------------------------------------------------------------------------
template <int E>
__global__ void gemm_wmma_k(const __bf16* __restrict__ A, const __bf16* __restrict__ Wt,
                            const float* __restrict__ bias,
                            float* __restrict__ Cf, __bf16* __restrict__ Cb,
                            int M, int N, int K) {
  const int lane = threadIdx.x;
  const int n0 = blockIdx.x * 64;
  const int m0 = blockIdx.y * 32;
  const int l15 = lane & 15;
  const int hi = lane >> 4;
  const __bf16* ap0 = A + (size_t)(m0 + l15) * K + hi * 8;
  const __bf16* ap1 = ap0 + (size_t)16 * K;
  const __bf16* bp = Wt + (size_t)(n0 + l15) * K + hi * 8;
  const size_t bs = (size_t)16 * K;
  f32x8 acc[2][4] = {};
  for (int k0 = 0; k0 < K; k0 += 32) {
    bf16x16 a0 = load_afrag(ap0 + k0);
    bf16x16 a1 = load_afrag(ap1 + k0);
    bf16x16 b0 = load_bfrag(bp + k0);
    bf16x16 b1 = load_bfrag(bp + bs + k0);
    bf16x16 b2 = load_bfrag(bp + 2 * bs + k0);
    bf16x16 b3 = load_bfrag(bp + 3 * bs + k0);
    acc[0][0] = wmma_bf16(a0, b0, acc[0][0]);
    acc[0][1] = wmma_bf16(a0, b1, acc[0][1]);
    acc[0][2] = wmma_bf16(a0, b2, acc[0][2]);
    acc[0][3] = wmma_bf16(a0, b3, acc[0][3]);
    acc[1][0] = wmma_bf16(a1, b0, acc[1][0]);
    acc[1][1] = wmma_bf16(a1, b1, acc[1][1]);
    acc[1][2] = wmma_bf16(a1, b2, acc[1][2]);
    acc[1][3] = wmma_bf16(a1, b3, acc[1][3]);
  }
#pragma unroll
  for (int mi = 0; mi < 2; ++mi) {
    const int r0 = m0 + mi * 16 + hi * 8;
#pragma unroll
    for (int j = 0; j < 4; ++j) {
      const int c = n0 + j * 16 + l15;
#pragma unroll
      for (int v = 0; v < 8; ++v) {
        const int r = r0 + v;
        const size_t idx = (size_t)r * N + c;
        float val = acc[mi][j][v];
        if constexpr (E == 0)      Cf[idx] = val;
        else if constexpr (E == 1) Cf[idx] = val + bias[c];
        else if constexpr (E == 2) Cf[idx] += val + bias[c];
        else if constexpr (E == 3) Cb[idx] = (__bf16)gelu_f(val + bias[c]);
        else if constexpr (E == 4) Cf[idx] = 2.0f * (val + bias[c]);
      }
    }
  }
}

// Attention scores: S = (q*scale) @ k^T + slope*ascale*(n-m); mask cols >= T.
// Per (b,h): M=N=512, K=64. grid=(8,16,48), block=32.
__global__ void attn_scores_k(const __bf16* __restrict__ Q, const __bf16* __restrict__ Kb,
                              float* __restrict__ S, const float* __restrict__ ascale,
                              int T) {
  const int lane = threadIdx.x;
  const int bh = blockIdx.z;
  const int h = bh % 12;
  const __bf16* A = Q  + (size_t)bh * 512 * 64;
  const __bf16* W = Kb + (size_t)bh * 512 * 64;
  float* Sp = S + (size_t)bh * 512 * 512;
  const int n0 = blockIdx.x * 64;
  const int m0 = blockIdx.y * 32;
  const int l15 = lane & 15;
  const int hi = lane >> 4;
  const __bf16* ap0 = A + (size_t)(m0 + l15) * 64 + hi * 8;
  const __bf16* ap1 = ap0 + (size_t)16 * 64;
  const __bf16* bp = W + (size_t)(n0 + l15) * 64 + hi * 8;
  const size_t bs = (size_t)16 * 64;
  f32x8 acc[2][4] = {};
#pragma unroll
  for (int k0 = 0; k0 < 64; k0 += 32) {
    bf16x16 a0 = load_afrag(ap0 + k0);
    bf16x16 a1 = load_afrag(ap1 + k0);
    bf16x16 b0 = load_bfrag(bp + k0);
    bf16x16 b1 = load_bfrag(bp + bs + k0);
    bf16x16 b2 = load_bfrag(bp + 2 * bs + k0);
    bf16x16 b3 = load_bfrag(bp + 3 * bs + k0);
    acc[0][0] = wmma_bf16(a0, b0, acc[0][0]);
    acc[0][1] = wmma_bf16(a0, b1, acc[0][1]);
    acc[0][2] = wmma_bf16(a0, b2, acc[0][2]);
    acc[0][3] = wmma_bf16(a0, b3, acc[0][3]);
    acc[1][0] = wmma_bf16(a1, b0, acc[1][0]);
    acc[1][1] = wmma_bf16(a1, b1, acc[1][1]);
    acc[1][2] = wmma_bf16(a1, b2, acc[1][2]);
    acc[1][3] = wmma_bf16(a1, b3, acc[1][3]);
  }
  const float slope = (h < 8) ? exp2f(-(float)(h + 1)) : exp2f(-((float)(h - 8) + 0.5f));
  const float as = slope * ascale[0];
#pragma unroll
  for (int mi = 0; mi < 2; ++mi) {
    const int r0 = m0 + mi * 16 + hi * 8;
#pragma unroll
    for (int j = 0; j < 4; ++j) {
      const int c = n0 + j * 16 + l15;
#pragma unroll
      for (int v = 0; v < 8; ++v) {
        const int r = r0 + v;
        float val = acc[mi][j][v] + as * (float)(c - r);
        if (c >= T) val = -1e30f;
        Sp[(size_t)r * 512 + c] = val;
      }
    }
  }
}

// o = att @ v. Per (b,h): M=512, N=64, K=512; Wt = v^T [64][512].
// Writes bf16 straight into gathered obf[(b*512+t)][h*64+d]. grid=(1,16,48).
__global__ void attn_o_k(const __bf16* __restrict__ Att, const __bf16* __restrict__ Vt,
                         __bf16* __restrict__ obf) {
  const int lane = threadIdx.x;
  const int bh = blockIdx.z;
  const int b = bh / 12, h = bh % 12;
  const __bf16* A = Att + (size_t)bh * 512 * 512;
  const __bf16* W = Vt  + (size_t)bh * 64 * 512;
  const int m0 = blockIdx.y * 32;
  const int l15 = lane & 15;
  const int hi = lane >> 4;
  const __bf16* ap0 = A + (size_t)(m0 + l15) * 512 + hi * 8;
  const __bf16* ap1 = ap0 + (size_t)16 * 512;
  const __bf16* bp = W + (size_t)l15 * 512 + hi * 8;
  const size_t bs = (size_t)16 * 512;
  f32x8 acc[2][4] = {};
  for (int k0 = 0; k0 < 512; k0 += 32) {
    bf16x16 a0 = load_afrag(ap0 + k0);
    bf16x16 a1 = load_afrag(ap1 + k0);
    bf16x16 b0 = load_bfrag(bp + k0);
    bf16x16 b1 = load_bfrag(bp + bs + k0);
    bf16x16 b2 = load_bfrag(bp + 2 * bs + k0);
    bf16x16 b3 = load_bfrag(bp + 3 * bs + k0);
    acc[0][0] = wmma_bf16(a0, b0, acc[0][0]);
    acc[0][1] = wmma_bf16(a0, b1, acc[0][1]);
    acc[0][2] = wmma_bf16(a0, b2, acc[0][2]);
    acc[0][3] = wmma_bf16(a0, b3, acc[0][3]);
    acc[1][0] = wmma_bf16(a1, b0, acc[1][0]);
    acc[1][1] = wmma_bf16(a1, b1, acc[1][1]);
    acc[1][2] = wmma_bf16(a1, b2, acc[1][2]);
    acc[1][3] = wmma_bf16(a1, b3, acc[1][3]);
  }
#pragma unroll
  for (int mi = 0; mi < 2; ++mi) {
    const int r0 = m0 + mi * 16 + hi * 8;
#pragma unroll
    for (int j = 0; j < 4; ++j) {
      const int c = j * 16 + l15;  // 0..63
#pragma unroll
      for (int v = 0; v < 8; ++v) {
        const int r = r0 + v;
        obf[((size_t)(b * 512 + r)) * 768 + h * 64 + c] = (__bf16)acc[mi][j][v];
      }
    }
  }
}

// ---------------- LayerNorm / softmax / elementwise kernels ----------------

// Affine row LN -> bf16. One wave per row.
__global__ void ln_row_bf16_k(const float* __restrict__ x, const float* __restrict__ g,
                              const float* __restrict__ b, __bf16* __restrict__ o, int C) {
  const int row = blockIdx.x, lane = threadIdx.x;
  const float* xr = x + (size_t)row * C;
  float s = 0.f, s2 = 0.f;
  for (int c = lane; c < C; c += 32) { float v = xr[c]; s += v; s2 += v * v; }
  s = wave_sum(s); s2 = wave_sum(s2);
  const float m = s / C;
  const float r = rsqrtf(s2 / C - m * m + 1e-5f);
  __bf16* orow = o + (size_t)row * C;
  for (int c = lane; c < C; c += 32) orow[c] = (__bf16)((xr[c] - m) * r * g[c] + b[c]);
}

// Non-affine row LN + gelu -> f32 (positional conv stack).
__global__ void ln_row_gelu_k(const float* __restrict__ x, float* __restrict__ o, int C) {
  const int row = blockIdx.x, lane = threadIdx.x;
  const float* xr = x + (size_t)row * C;
  float s = 0.f, s2 = 0.f;
  for (int c = lane; c < C; c += 32) { float v = xr[c]; s += v; s2 += v * v; }
  s = wave_sum(s); s2 = wave_sum(s2);
  const float m = s / C;
  const float r = rsqrtf(s2 / C - m * m + 1e-5f);
  float* orow = o + (size_t)row * C;
  for (int c = lane; c < C; c += 32) orow[c] = gelu_f((xr[c] - m) * r);
}

// Final affine LN, write only valid tokens to d_out [4][499][768].
__global__ void final_ln_k(const float* __restrict__ x, const float* __restrict__ g,
                           const float* __restrict__ b, float* __restrict__ out, int T) {
  const int row = blockIdx.x, lane = threadIdx.x;
  const int bb = row >> 9, t = row & 511;
  if (t >= T) return;
  const float* xr = x + (size_t)row * 768;
  float s = 0.f, s2 = 0.f;
  for (int c = lane; c < 768; c += 32) { float v = xr[c]; s += v; s2 += v * v; }
  s = wave_sum(s); s2 = wave_sum(s2);
  const float m = s / 768.f;
  const float r = rsqrtf(s2 / 768.f - m * m + 1e-5f);
  float* orow = out + ((size_t)(bb * T + t)) * 768;
  for (int c = lane; c < 768; c += 32) orow[c] = (xr[c] - m) * r * g[c] + b[c];
}

// Conv-stack channel LN + gelu over [512][Talloc] (per time step). In-place ok.
__global__ void conv_ln_gelu_k(const float* __restrict__ in, float* __restrict__ out,
                               const float* __restrict__ g, const float* __restrict__ b,
                               int Talloc, int Tlog) {
  const int t = blockIdx.x * 256 + threadIdx.x;
  if (t >= Tlog) return;
  float s = 0.f, s2 = 0.f;
  for (int c = 0; c < 512; ++c) { float v = in[(size_t)c * Talloc + t]; s += v; s2 += v * v; }
  const float m = s * (1.f / 512.f);
  const float r = rsqrtf(s2 * (1.f / 512.f) - m * m + 1e-5f);
  for (int c = 0; c < 512; ++c) {
    float v = (in[(size_t)c * Talloc + t] - m) * r * g[c] + b[c];
    out[(size_t)c * Talloc + t] = gelu_f(v);
  }
}

// First conv layer (Cin=1, k=10, s=5). Writes zeros in padded tail.
__global__ void conv0_k(const float* __restrict__ src, const float* __restrict__ w,
                        float* __restrict__ out, int Tout, int Tp) {
  const int idx = blockIdx.x * 256 + threadIdx.x;
  if (idx >= 512 * Tp) return;
  const int c = idx / Tp, t = idx % Tp;
  if (t >= Tout) { out[idx] = 0.f; return; }
  const float* wp = w + c * 10;
  const float* sp = src + 5 * t;
  float s = 0.f;
#pragma unroll
  for (int j = 0; j < 10; ++j) s += wp[j] * sp[j];
  out[idx] = s;
}

// im2col -> bf16 [Tp][Cin*k] for stride-2 conv layers. Zero pads everything OOB.
__global__ void im2col_k(const float* __restrict__ in, __bf16* __restrict__ out,
                         int ksz, int TinA, int TinLog, int ToutLog, int Tp, int Kd) {
  const size_t idx = (size_t)blockIdx.x * 256 + threadIdx.x;
  if (idx >= (size_t)Tp * Kd) return;
  const int t = (int)(idx / Kd), q = (int)(idx % Kd);
  const int ci = q / ksz, j = q % ksz;
  const int tin = t * 2 + j;
  float v = 0.f;
  if (t < ToutLog && tin < TinLog) v = in[(size_t)ci * TinA + tin];
  out[idx] = (__bf16)v;
}

// Grouped positional conv (k=19, pad=9, groups=16) + bias, bounds-checked at T.
__global__ void pos_conv_k(const float* __restrict__ p, const float* __restrict__ w,
                           const float* __restrict__ bias, float* __restrict__ out, int T) {
  const int idx = blockIdx.x * 256 + threadIdx.x;
  if (idx >= 2048 * 768) return;
  const int r = idx / 768, c = idx % 768;
  const int b = r >> 9, t = r & 511;
  const int cbase = (c / 48) * 48;
  const float* wr = w + (size_t)c * 48 * 19;
  float s = bias[c];
  for (int j = 0; j < 19; ++j) {
    const int tt = t + j - 9;
    if (tt < 0 || tt >= T) continue;
    const float* pr = p + ((size_t)(b * 512 + tt)) * 768 + cbase;
#pragma unroll
    for (int ci = 0; ci < 48; ++ci) s += wr[ci * 19 + j] * pr[ci];
  }
  out[idx] = s;
}

__global__ void add_k(const float* __restrict__ a, const float* __restrict__ b,
                      float* __restrict__ o, int n) {
  const int i = blockIdx.x * 256 + threadIdx.x;
  if (i < n) o[i] = a[i] + b[i];
}

// Split qkv -> q*scale (bf16), k (bf16), v^T (bf16), zero-padded past T.
__global__ void qkv_split_k(const float* __restrict__ qkv, __bf16* __restrict__ qb,
                            __bf16* __restrict__ kb, __bf16* __restrict__ vt, int T) {
  const int idx = blockIdx.x * 256 + threadIdx.x;
  if (idx >= 48 * 512 * 64) return;
  const int d = idx & 63;
  const int t = (idx >> 6) & 511;
  const int bh = idx >> 15;
  const int b = bh / 12, h = bh % 12;
  float q = 0.f, k = 0.f, v = 0.f;
  if (t < T) {
    const size_t base = ((size_t)(b * 512 + t)) * 2304 + h * 64 + d;
    q = qkv[base] * 0.125f;          // (DIM/HEADS)^-0.5 = 64^-0.5
    k = qkv[base + 768];
    v = qkv[base + 1536];
  }
  qb[((size_t)bh * 512 + t) * 64 + d] = (__bf16)q;
  kb[((size_t)bh * 512 + t) * 64 + d] = (__bf16)k;
  vt[((size_t)bh * 64 + d) * 512 + t] = (__bf16)v;
}

// Row softmax over 512 cols -> bf16. One wave per row.
__global__ void softmax_k(const float* __restrict__ S, __bf16* __restrict__ att) {
  const int row = blockIdx.x, lane = threadIdx.x;
  const float* sr = S + (size_t)row * 512;
  float e[16];
  float mx = -3.4e38f;
#pragma unroll
  for (int j = 0; j < 16; ++j) mx = fmaxf(mx, sr[lane + 32 * j]);
  mx = wave_max(mx);
  float sum = 0.f;
#pragma unroll
  for (int j = 0; j < 16; ++j) { e[j] = __expf(sr[lane + 32 * j] - mx); sum += e[j]; }
  sum = wave_sum(sum);
  const float inv = 1.0f / sum;
  __bf16* ar = att + (size_t)row * 512;
#pragma unroll
  for (int j = 0; j < 16; ++j) ar[lane + 32 * j] = (__bf16)(e[j] * inv);
}

// Pack weight [K][N] f32 -> [N][K] bf16 (WMMA B-operand friendly).
__global__ void pack_wt_k(const float* __restrict__ W, __bf16* __restrict__ o,
                          int K, int N) {
  const size_t idx = (size_t)blockIdx.x * 256 + threadIdx.x;
  if (idx >= (size_t)N * K) return;
  const int n = (int)(idx / K), kk = (int)(idx % K);
  o[idx] = (__bf16)W[(size_t)kk * N + n];
}

__global__ void f2bf_k(const float* __restrict__ in, __bf16* __restrict__ o, int n) {
  const int i = blockIdx.x * 256 + threadIdx.x;
  if (i < n) o[i] = (__bf16)in[i];
}

// Extractor LN over 512 channels + transpose -> xln[2048][512] bf16 (pad rows 0).
__global__ void extract_ln_k(const float* __restrict__ fb, const float* __restrict__ g,
                             const float* __restrict__ b, __bf16* __restrict__ xln, int T) {
  const int idx = blockIdx.x * 256 + threadIdx.x;
  if (idx >= 2048) return;
  const int bb = idx >> 9, t = idx & 511;
  __bf16* orow = xln + (size_t)idx * 512;
  if (t >= T) { for (int c = 0; c < 512; ++c) orow[c] = (__bf16)0.f; return; }
  const float* base = fb + (size_t)bb * 512 * 512 + t;
  float s = 0.f, s2 = 0.f;
  for (int c = 0; c < 512; ++c) { float v = base[(size_t)c * 512]; s += v; s2 += v * v; }
  const float m = s * (1.f / 512.f);
  const float r = rsqrtf(s2 * (1.f / 512.f) - m * m + 1e-5f);
  for (int c = 0; c < 512; ++c) orow[c] = (__bf16)((base[(size_t)c * 512] - m) * r * g[c] + b[c]);
}

// ---------------------------------------------------------------------------

extern "C" void kernel_launch(void* const* d_in, const int* in_sizes, int n_in,
                              void* d_out, int out_size, void* d_ws, size_t ws_size,
                              hipStream_t stream) {
  (void)in_sizes; (void)n_in; (void)out_size; (void)ws_size;

  // ---- input pointers (setup_inputs flatten order) ----
  const float* src = (const float*)d_in[0];
  const float* convw[7]; const float* convg[7]; const float* convb[7];
  for (int i = 0; i < 7; ++i) {
    convw[i] = (const float*)d_in[1 + 3 * i];
    convg[i] = (const float*)d_in[2 + 3 * i];
    convb[i] = (const float*)d_in[3 + 3 * i];
  }
  const float* exg = (const float*)d_in[22];
  const float* exb = (const float*)d_in[23];
  const float* proj_w = (const float*)d_in[24];
  const float* proj_b = (const float*)d_in[25];
  const float* posw[5]; const float* posb[5];
  for (int i = 0; i < 5; ++i) {
    posw[i] = (const float*)d_in[26 + 2 * i];
    posb[i] = (const float*)d_in[27 + 2 * i];
  }
  const int BLK0 = 36;  // 12 entries per block
  const float* fing = (const float*)d_in[180];
  const float* finb = (const float*)d_in[181];
  const float* ascale = (const float*)d_in[182];

  // ---- workspace carve-up ----
  char* base = (char*)d_ws;
  size_t off = 0;
  auto alloc = [&](size_t bytes) -> void* {
    void* p = base + off;
    off = (off + bytes + 255) & ~(size_t)255;
    return p;
  };
  __bf16* cwbf[6];
  const int cK[6] = {1536, 1536, 1536, 1536, 1024, 1024};
  for (int l = 0; l < 6; ++l) cwbf[l] = (__bf16*)alloc((size_t)512 * cK[l] * 2);
  float*  cbufA   = (float*)alloc((size_t)512 * 32000 * 4);
  float*  cbufB   = (float*)alloc((size_t)512 * 16000 * 4);
  __bf16* colbuf  = (__bf16*)alloc((size_t)16000 * 1536 * 2);
  float*  finalbf = (float*)alloc((size_t)4 * 512 * 512 * 4);
  __bf16* xln     = (__bf16*)alloc((size_t)2048 * 512 * 2);
  __bf16* projWt  = (__bf16*)alloc((size_t)768 * 512 * 2);
  float*  x0      = (float*)alloc((size_t)2048 * 768 * 4);
  float*  pA      = (float*)alloc((size_t)2048 * 768 * 4);
  float*  pB      = (float*)alloc((size_t)2048 * 768 * 4);
  float*  xbuf    = (float*)alloc((size_t)2048 * 768 * 4);
  __bf16* hbf     = (__bf16*)alloc((size_t)2048 * 768 * 2);
  __bf16* mbf     = (__bf16*)alloc((size_t)2048 * 768 * 2);
  __bf16* obf     = (__bf16*)alloc((size_t)2048 * 768 * 2);
  __bf16* h1bf    = (__bf16*)alloc((size_t)2048 * 3072 * 2);
  __bf16* qkvWt   = (__bf16*)alloc((size_t)2304 * 768 * 2);
  __bf16* oprojWt = (__bf16*)alloc((size_t)768 * 768 * 2);
  __bf16* fc1Wt   = (__bf16*)alloc((size_t)3072 * 768 * 2);
  __bf16* fc2Wt   = (__bf16*)alloc((size_t)768 * 3072 * 2);
  float*  qkvbuf  = (float*)alloc((size_t)2048 * 2304 * 4);
  __bf16* qbf     = (__bf16*)alloc((size_t)48 * 512 * 64 * 2);
  __bf16* kbf     = (__bf16*)alloc((size_t)48 * 512 * 64 * 2);
  __bf16* vtbf    = (__bf16*)alloc((size_t)48 * 64 * 512 * 2);
  float*  sbuf    = (float*)alloc((size_t)48 * 512 * 512 * 4);
  __bf16* attbf   = (__bf16*)alloc((size_t)48 * 512 * 512 * 2);

  const int T = 499;

  // ---- weight prep (conv + projection; per-block weights packed in the loop) ----
  for (int l = 0; l < 6; ++l) {
    const int n = 512 * cK[l];
    f2bf_k<<<(n + 255) / 256, 256, 0, stream>>>(convw[l + 1], cwbf[l], n);
  }
  {
    const size_t n = (size_t)768 * 512;
    pack_wt_k<<<(unsigned)((n + 255) / 256), 256, 0, stream>>>(proj_w, projWt, 512, 768);
  }

  // ---- conv feature extractor (per batch element) ----
  struct CL { int k, Tin, TinA, Tout, Tp, K; };
  const CL cls[6] = {
    {3, 31999, 32000, 15999, 16000, 1536},
    {3, 15999, 16000,  7999,  8000, 1536},
    {3,  7999,  8000,  3999,  4032, 1536},
    {3,  3999,  4032,  1999,  2048, 1536},
    {2,  1999,  2048,   999,  1024, 1024},
    {2,   999,  1024,   499,   512, 1024},
  };
  for (int b = 0; b < 4; ++b) {
    conv0_k<<<(512 * 32000) / 256, 256, 0, stream>>>(src + (size_t)b * 160000, convw[0],
                                                     cbufA, 31999, 32000);
    conv_ln_gelu_k<<<(31999 + 255) / 256, 256, 0, stream>>>(cbufA, cbufA, convg[0],
                                                            convb[0], 32000, 31999);
    float* bufs[2] = {cbufA, cbufB};
    int cur = 0;
    for (int l = 0; l < 6; ++l) {
      const CL& L = cls[l];
      float* in = bufs[cur];
      float* out = (l == 5) ? (finalbf + (size_t)b * 512 * 512) : bufs[cur ^ 1];
      const size_t nc = (size_t)L.Tp * L.K;
      im2col_k<<<(unsigned)((nc + 255) / 256), 256, 0, stream>>>(in, colbuf, L.k, L.TinA,
                                                                 L.Tin, L.Tout, L.Tp, L.K);
      gemm_wmma_k<0><<<dim3(L.Tp / 64, 512 / 32), 32, 0, stream>>>(
          cwbf[l], colbuf, nullptr, out, nullptr, 512, L.Tp, L.K);
      conv_ln_gelu_k<<<(L.Tout + 255) / 256, 256, 0, stream>>>(out, out, convg[l + 1],
                                                               convb[l + 1], L.Tp, L.Tout);
      cur ^= 1;
    }
  }

  // ---- extractor LN + transpose -> bf16, then projection to 768 ----
  extract_ln_k<<<8, 256, 0, stream>>>(finalbf, exg, exb, xln, T);
  gemm_wmma_k<1><<<dim3(768 / 64, 2048 / 32), 32, 0, stream>>>(
      xln, projWt, proj_b, x0, nullptr, 2048, 768, 512);

  // ---- positional conv stack ----
  const float* pin = x0;
  for (int d = 0; d < 5; ++d) {
    pos_conv_k<<<(2048 * 768 + 255) / 256, 256, 0, stream>>>(pin, posw[d], posb[d], pA, T);
    ln_row_gelu_k<<<2048, 32, 0, stream>>>(pA, pB, 768);
    pin = pB;
  }
  add_k<<<(2048 * 768 + 255) / 256, 256, 0, stream>>>(x0, pB, xbuf, 2048 * 768);

  // ---- transformer blocks ----
  for (int i = 0; i < 12; ++i) {
    const int B0 = BLK0 + 12 * i;
    const float* n1g = (const float*)d_in[B0 + 0];
    const float* n1b = (const float*)d_in[B0 + 1];
    const float* qkv_w = (const float*)d_in[B0 + 2];
    const float* qkv_b = (const float*)d_in[B0 + 3];
    const float* pw = (const float*)d_in[B0 + 4];
    const float* pb = (const float*)d_in[B0 + 5];
    const float* n2g = (const float*)d_in[B0 + 6];
    const float* n2b = (const float*)d_in[B0 + 7];
    const float* fc1w = (const float*)d_in[B0 + 8];
    const float* fc1b = (const float*)d_in[B0 + 9];
    const float* fc2w = (const float*)d_in[B0 + 10];
    const float* fc2b = (const float*)d_in[B0 + 11];

    pack_wt_k<<<(768 * 2304 + 255) / 256, 256, 0, stream>>>(qkv_w, qkvWt, 768, 2304);
    pack_wt_k<<<(768 * 768 + 255) / 256, 256, 0, stream>>>(pw, oprojWt, 768, 768);
    pack_wt_k<<<(768 * 3072 + 255) / 256, 256, 0, stream>>>(fc1w, fc1Wt, 768, 3072);
    pack_wt_k<<<(3072 * 768 + 255) / 256, 256, 0, stream>>>(fc2w, fc2Wt, 3072, 768);

    ln_row_bf16_k<<<2048, 32, 0, stream>>>(xbuf, n1g, n1b, hbf, 768);
    gemm_wmma_k<1><<<dim3(2304 / 64, 2048 / 32), 32, 0, stream>>>(
        hbf, qkvWt, qkv_b, qkvbuf, nullptr, 2048, 2304, 768);
    qkv_split_k<<<(48 * 512 * 64 + 255) / 256, 256, 0, stream>>>(qkvbuf, qbf, kbf, vtbf, T);
    attn_scores_k<<<dim3(8, 16, 48), 32, 0, stream>>>(qbf, kbf, sbuf, ascale, T);
    softmax_k<<<48 * 512, 32, 0, stream>>>(sbuf, attbf);
    attn_o_k<<<dim3(1, 16, 48), 32, 0, stream>>>(attbf, vtbf, obf);
    gemm_wmma_k<2><<<dim3(768 / 64, 2048 / 32), 32, 0, stream>>>(
        obf, oprojWt, pb, xbuf, nullptr, 2048, 768, 768);  // x += o@W + b
    ln_row_bf16_k<<<2048, 32, 0, stream>>>(xbuf, n2g, n2b, mbf, 768);
    gemm_wmma_k<3><<<dim3(3072 / 64, 2048 / 32), 32, 0, stream>>>(
        mbf, fc1Wt, fc1b, nullptr, h1bf, 2048, 3072, 768);  // gelu -> bf16
    gemm_wmma_k<4><<<dim3(768 / 64, 2048 / 32), 32, 0, stream>>>(
        h1bf, fc2Wt, fc2b, xbuf, nullptr, 2048, 768, 3072); // x = m + m
  }

  // ---- final LN -> output ----
  final_ln_k<<<2048, 32, 0, stream>>>(xbuf, fing, finb, (float*)d_out, T);
}
